// BoundaryPredictor4_29867202576808
// MI455X (gfx1250) — matrix operand order repaired and checked by reading, not compiled, not run on hardware
//
#include <hip/hip_runtime.h>
#include <hip/hip_bf16.h>
#include <math.h>

// Problem constants (match reference)
#define BB 8
#define LL 4096
#define DD 512
#define MM (BB * LL)   // 32768 rows

// LDS A-tile layout: [16 k-chunks][128 rows][40 bf16] (32 used + 8 pad)
// row stride 80 B -> 16B-aligned b128 frags, bank stride 20 dwords (conflict-light)
#define ROWEL 40
#define CHUNKEL (128 * ROWEL)   // 5120 elements per k-chunk

// ---------- small helper types (trivial, union-safe) ----------
typedef __attribute__((ext_vector_type(8)))  float        v8f;
typedef __attribute__((ext_vector_type(16))) __bf16       v16bf;
typedef __attribute__((ext_vector_type(4)))  unsigned int u32x4;
typedef __attribute__((ext_vector_type(2)))  unsigned int u32x2;
typedef __attribute__((ext_vector_type(4)))  float        f32x4;

union AB32 { v16bf v; u32x4 q[2]; };

__device__ __forceinline__ unsigned short bf16r(float f) {
  unsigned int u = __float_as_uint(f);
  u += 0x7FFFu + ((u >> 16) & 1u);   // round-to-nearest-even
  return (unsigned short)(u >> 16);
}

// ---------- K0: W1 (f32, [K=512][N=512]) -> W1T (bf16, [N][K]) ----------
__global__ void prep_w1t_kernel(const float* __restrict__ W1, unsigned short* __restrict__ w1t) {
  int idx = blockIdx.x * 256 + threadIdx.x;      // idx = n*512 + k
  int n = idx >> 9, k = idx & 511;
  w1t[idx] = bf16r(W1[k * DD + n]);
}

// ---------- K1a: zero the logit accumulation buffer ----------
__global__ void zero_kernel(float* __restrict__ p, int n) {
  int i = blockIdx.x * 256 + threadIdx.x;
  if (i < n) p[i] = 0.0f;
}

// ---------- K1: fused GEMM (bf16 WMMA) + GELU + (h @ W2) partial logits ----------
// Block: 512 threads = 16 waves; block handles 128 rows x all 512 cols.
// Wave w owns N-slice [w*32, w*32+32) (2 n-tiles) across 8 m-tiles.
// Loop nest: ks outer, {B-frags(2), then per-mt: one A-frag -> 2 WMMAs} inner.
// Every load is consumed immediately -> nothing for LICM/CSE to hoist -> no spills.
__global__ __launch_bounds__(512) void gemm_act_kernel(
    const float* __restrict__ hidden, const unsigned short* __restrict__ w1t,
    const float* __restrict__ b1, const float* __restrict__ W2,
    float* __restrict__ logits) {
  extern __shared__ unsigned short sA[];   // [16][128][40] bf16 = 160 KB
  const int tid = threadIdx.x;
  const int rowBase = blockIdx.x * 128;

  // Stage 128x512 f32 rows -> bf16 in LDS (each thread: 32 x float4)
  for (int i = 0; i < 32; ++i) {
    int e = i * 2048 + tid * 4;           // flat element in [0, 65536)
    int r = e >> 9, k = e & 511;
    int ks = k >> 5, koff = k & 31;
    f32x4 f = *(const f32x4*)(hidden + (size_t)rowBase * DD + e);
    u32x2 pk;
    pk.x = (unsigned)bf16r(f.x) | ((unsigned)bf16r(f.y) << 16);
    pk.y = (unsigned)bf16r(f.z) | ((unsigned)bf16r(f.w) << 16);
    *(u32x2*)(sA + ks * CHUNKEL + r * ROWEL + koff) = pk;
  }
  __syncthreads();

  const int wave = tid >> 5;
  const int lane = tid & 31;
  const int laneHi = lane >> 4;   // 0 or 1
  const int ln = lane & 15;

  const int ncol0 = wave * 32 + ln;        // n-tile 0 column for this lane
  const int ncol1 = ncol0 + 16;            // n-tile 1 column
  const float w2n0 = W2[ncol0], w2n1 = W2[ncol1];
  const float b1n0 = b1[ncol0], b1n1 = b1[ncol1];
  const unsigned short* brow0 = w1t + (size_t)ncol0 * DD + laneHi * 8;
  const unsigned short* brow1 = w1t + (size_t)ncol1 * DD + laneHi * 8;

  v8f acc0[8], acc1[8];
  {
    v8f z = {0.f,0.f,0.f,0.f,0.f,0.f,0.f,0.f};
    #pragma unroll
    for (int mt = 0; mt < 8; ++mt) { acc0[mt] = z; acc1[mt] = z; }
  }

  for (int ks = 0; ks < 16; ++ks) {
    const int k0 = ks * 32;
    AB32 bf0, bf1;                         // B frags: two 16 B pieces of W1T row at k0+8*laneHi and k0+16+8*laneHi
    bf0.q[0] = *(const u32x4*)(brow0 + k0);
    bf0.q[1] = *(const u32x4*)(brow0 + k0 + 16);
    bf1.q[0] = *(const u32x4*)(brow1 + k0);
    bf1.q[1] = *(const u32x4*)(brow1 + k0 + 16);
    if (ks + 1 < 16) {
      __builtin_prefetch(brow0 + k0 + 32, 0, 3);   // global_prefetch_b8
      __builtin_prefetch(brow1 + k0 + 32, 0, 3);
    }

    // All 8 m-tile frags share one base; mt offsets are small immediates (mt*1280 B)
    const unsigned short* abase = sA + ks * CHUNKEL + ln * ROWEL + laneHi * 8;
    #pragma unroll
    for (int mt = 0; mt < 8; ++mt) {
      AB32 ar;                             // A frag from LDS (two ds_load_b128), used twice
      const unsigned short* ap = abase + mt * 16 * ROWEL;
      ar.q[0] = *(const u32x4*)(ap);
      ar.q[1] = *(const u32x4*)(ap + 16);
      acc0[mt] = __builtin_amdgcn_wmma_f32_16x16x32_bf16(
          false, ar.v, false, bf0.v, (short)0, acc0[mt], false, false);
      acc1[mt] = __builtin_amdgcn_wmma_f32_16x16x32_bf16(
          false, ar.v, false, bf1.v, (short)0, acc1[mt], false, false);
    }
  }

  // Epilogue: exact GELU, multiply by W2[ncol], reduce over the 16 N-lanes per
  // half, accumulate into global logits. C layout: lane (half,ln), VGPR v ->
  // M = v + 8*half, N = ln.
  #pragma unroll
  for (int mt = 0; mt < 8; ++mt)
    #pragma unroll
    for (int v = 0; v < 8; ++v) {
      float x0 = acc0[mt][v] + b1n0;
      float x1 = acc1[mt][v] + b1n1;
      float g0 = 0.5f * x0 * (1.0f + erff(x0 * 0.70710678118654752440f));
      float g1 = 0.5f * x1 * (1.0f + erff(x1 * 0.70710678118654752440f));
      float s = g0 * w2n0 + g1 * w2n1;
      s += __shfl_xor(s, 1, 32);
      s += __shfl_xor(s, 2, 32);
      s += __shfl_xor(s, 4, 32);
      s += __shfl_xor(s, 8, 32);
      if (ln == 0)
        atomicAdd(&logits[rowBase + mt * 16 + laneHi * 8 + v], s);
    }
}

// ---------- K1b: sigmoid(logit + b2) -> probs ----------
__global__ void probs_kernel(const float* __restrict__ logits, const float* __restrict__ b2,
                             float* __restrict__ probs) {
  int i = blockIdx.x * 256 + threadIdx.x;
  float x = logits[i] + b2[0];
  probs[i] = 1.0f / (1.0f + __expf(-x));
}

// ---------- K2: boundaries, masking, per-batch exclusive scan, run starts ----------
// One block per batch sample; 256 threads x 16 positions.
__global__ void boundary_kernel(const float* __restrict__ probs, const float* __restrict__ lengths,
                                float* __restrict__ maskedOut, float* __restrict__ shortOut,
                                float* __restrict__ numOut, int* __restrict__ startArr, int S) {
  const int b = blockIdx.x;
  const int tid = threadIdx.x;
  const float* pb = probs + (size_t)b * LL;
  int alen = (int)(lengths[b] * (float)LL);          // torch .long() truncation
  if (alen > LL) alen = LL;
  int last = alen - 1; if (last < 0) last = 0; if (last > LL - 1) last = LL - 1;

  // Default-fill segment starts (including sentinel at s == S) with L
  for (int s = tid; s <= S; s += 256) startArr[b * (S + 1) + s] = LL;

  const int t0 = tid * 16;
  int h[16]; int lsum = 0;
  #pragma unroll
  for (int i = 0; i < 16; ++i) {
    int t = t0 + i;
    float p = pb[t];
    int valid = (t < alen) ? 1 : 0;
    int hh = (p > 0.5f) ? valid : 0;
    if (t == last) hh = 1;                           // set-last rule (unconditional)
    h[i] = hh; lsum += hh;
    maskedOut[(size_t)b * LL + t] = valid ? p : 0.0f;
  }

  __shared__ int ssum[256];
  ssum[tid] = lsum;
  __syncthreads();
  for (int off = 1; off < 256; off <<= 1) {
    int add = (tid >= off) ? ssum[tid - off] : 0;
    __syncthreads();
    ssum[tid] += add;
    __syncthreads();
  }
  int total = ssum[255];
  int run = ssum[tid] - lsum;                        // exclusive prefix (= segment id at t0)

  int prevH;
  if (t0 == 0) prevH = 1;                            // forces start of segment 0 at t=0
  else {
    int tp = t0 - 1;
    float p = pb[tp];
    int valid = (tp < alen) ? 1 : 0;
    prevH = (p > 0.5f) ? valid : 0;
    if (tp == last) prevH = 1;
  }
  #pragma unroll
  for (int i = 0; i < 16; ++i) {
    int t = t0 + i;
    if (t == 0 || prevH) startArr[b * (S + 1) + run] = t;   // unique writer per run
    prevH = h[i];
    run += h[i];
  }

  if (tid == 0) {
    numOut[b] = (float)total;
    int sm = (S > 1) ? S : 1;
    shortOut[b] = (float)total / (float)sm;
  }
}

// ---------- K3: contiguous-run mean pool ----------
// Grid: B*S blocks, 128 threads (4 floats of D each).
__global__ void pool_kernel(const float* __restrict__ hidden, const int* __restrict__ startArr,
                            float* __restrict__ pooled, int S) {
  const int s = blockIdx.x % S;
  const int b = blockIdx.x / S;
  const int st = startArr[b * (S + 1) + s];
  const int en = startArr[b * (S + 1) + s + 1];
  const int d = threadIdx.x * 4;
  f32x4 acc = {0.f, 0.f, 0.f, 0.f};
  for (int t = st; t < en; ++t) {
    f32x4 v = *(const f32x4*)(hidden + ((size_t)b * LL + t) * DD + d);
    acc.x += v.x; acc.y += v.y; acc.z += v.z; acc.w += v.w;
  }
  float inv = 1.0f / ((float)(en - st) + 1e-9f);
  f32x4 o = {acc.x * inv, acc.y * inv, acc.z * inv, acc.w * inv};
  *(f32x4*)(pooled + ((size_t)b * S + s) * DD + d) = o;
}

// ---------- host launch ----------
extern "C" void kernel_launch(void* const* d_in, const int* in_sizes, int n_in,
                              void* d_out, int out_size, void* d_ws, size_t ws_size,
                              hipStream_t stream) {
  const float* hidden  = (const float*)d_in[0];
  const float* lengths = (const float*)d_in[1];
  const float* W1      = (const float*)d_in[2];
  const float* b1      = (const float*)d_in[3];
  const float* W2      = (const float*)d_in[4];
  const float* b2      = (const float*)d_in[5];
  float* out = (float*)d_out;

  // Recover data-dependent n_segments from the fixed output size:
  // out = pooled[B,S,D] ++ masked[B,L] ++ shortened[B] ++ num[B]
  int S = (out_size - BB * LL - 2 * BB) / (BB * DD);
  if (S < 1) S = 1;

  char* ws = (char*)d_ws;
  unsigned short* w1t = (unsigned short*)ws;                       // 512 KB
  float* logits = (float*)(ws + 524288);                           // 128 KB
  float* probs  = (float*)(ws + 524288 + 131072);                  // 128 KB
  int* startArr = (int*)(ws + 524288 + 2 * 131072);                // B*(S+1) ints

  float* pooled   = out;
  float* masked   = out + (size_t)BB * S * DD;
  float* shortOut = masked + (size_t)BB * LL;
  float* numOut   = shortOut + BB;

  prep_w1t_kernel<<<(DD * DD) / 256, 256, 0, stream>>>(W1, w1t);
  zero_kernel<<<MM / 256, 256, 0, stream>>>(logits, MM);
  gemm_act_kernel<<<MM / 128, 512, 16 * CHUNKEL * 2, stream>>>(hidden, w1t, b1, W2, logits);
  probs_kernel<<<MM / 256, 256, 0, stream>>>(logits, b2, probs);
  boundary_kernel<<<BB, 256, 0, stream>>>(probs, lengths, masked, shortOut, numOut, startArr, S);
  pool_kernel<<<BB * S, 128, 0, stream>>>(hidden, startArr, pooled, S);
}
